// VQWeightDehazeNet_76416058130944
// MI455X (gfx1250) — matrix operand order repaired and checked by reading, not compiled, not run on hardware
//
#include <hip/hip_runtime.h>

// ---------------------------------------------------------------------------
// VQ-VAE vector quantization for MI455X (gfx1250, wave32, WMMA).
//   z:        [16, 256, 64, 64] fp32 (NCHW)   -> zf [N=65536, C=256]
//   codebook: [1024, 256] fp32
// Output: zq (NCHW, 16.7M floats) ++ scalar loss (1 float).
//
// argmin_e( ||e||^2 - 2 z.e ) via bf16 WMMA (f32 accum); exact fp32 gather +
// loss afterwards.  ||z||^2 dropped (row-constant, argmin-invariant).
// Each wave owns 32 rows (2 A sub-tiles in registers): every B fragment from
// LDS feeds TWO WMMAs.  B chunks are double-buffered in LDS and filled with
// GLOBAL_LOAD_ASYNC_TO_LDS_B128 (ASYNCcnt) so staging overlaps the WMMAs.
// ---------------------------------------------------------------------------

typedef __bf16 v16bf __attribute__((ext_vector_type(16)));
typedef float  v8f   __attribute__((ext_vector_type(8)));
typedef float  f32x4 __attribute__((ext_vector_type(4)));

#define N_E       1024
#define E_DIM     256
#define HW        4096          // 64*64
#define OUT_ELEMS 16777216      // 16*256*64*64
#define M_TILE    256           // spatial rows per workgroup (32 per wave)
#define E_CHUNK   64            // codewords staged in LDS per buffer
#define N_CHUNKS  (N_E / E_CHUNK)   // 16
#define K_CHUNKS  8             // 256 / 32

__device__ __forceinline__ void wait_asynccnt0() {
#if defined(__has_builtin) && __has_builtin(__builtin_amdgcn_s_wait_asynccnt)
  __builtin_amdgcn_s_wait_asynccnt(0);
#else
  asm volatile("s_wait_asynccnt 0x0" ::: "memory");
#endif
}

// ---------------------------------------------------------------------------
// Kernel 0: codebook fp32 -> bf16 (row-major), ||e||^2 fp32, zero loss slot.
// ---------------------------------------------------------------------------
__global__ __launch_bounds__(256) void vq_prep(const float* __restrict__ cb,
                                               __bf16* __restrict__ cb_bf,
                                               float* __restrict__ cbnorm,
                                               float* __restrict__ loss_out) {
  const int e = blockIdx.x;
  const int t = threadIdx.x;
  float v = cb[e * E_DIM + t];
  cb_bf[e * E_DIM + t] = (__bf16)v;
  float s = v * v;
  #pragma unroll
  for (int off = 16; off >= 1; off >>= 1) s += __shfl_down(s, off, 32);
  __shared__ float red[8];
  const int lane = t & 31, wv = t >> 5;
  if (lane == 0) red[wv] = s;
  __syncthreads();
  if (t == 0) {
    float tot = 0.f;
    #pragma unroll
    for (int i = 0; i < 8; ++i) tot += red[i];
    cbnorm[e] = tot;
    if (e == 0) *loss_out = 0.f;   // runs before main kernel (same stream)
  }
}

// ---------------------------------------------------------------------------
// Main kernel: 256 blocks x 256 threads (8 waves), 256 rows per block.
// ---------------------------------------------------------------------------
__global__ __launch_bounds__(256) void vq_main(const float* __restrict__ z,
                                               const float* __restrict__ cb_f32,
                                               const __bf16* __restrict__ cb_bf,
                                               const float* __restrict__ cbnorm,
                                               float* __restrict__ out,
                                               float* __restrict__ loss) {
  // Double-buffered B chunk, fragment-ordered:
  // [buf][e-tile][k-chunk][lane][16 bf16] = 2 x 32 KB
  __shared__ __align__(16) __bf16 bTile[2][E_CHUNK / 16][K_CHUNKS][32][16];
  __shared__ float cbnormS[N_E];          // 4 KB: ||e||^2, LDS-resident
  __shared__ int   idxArr[M_TILE];
  __shared__ float redBuf[8];

  const int t    = threadIdx.x;
  const int lane = t & 31;
  const int wv   = t >> 5;        // wave 0..7
  const int half = lane >> 4;     // hi/lo 16-lane group
  const int n16  = lane & 15;

  const int tile = blockIdx.x * M_TILE;          // 4096 % 256 == 0
  const int b    = tile >> 12;
  const int hwb  = tile & (HW - 1);
  const float* zplane = z + (size_t)b * E_DIM * HW + hwb;

  // Async staging of one 64-codeword chunk directly into LDS (no VGPRs).
  // B 32x16 layout: lane col n = lane&15; elems 0..7 -> K = kc*32+half*8+j,
  //                 elems 8..15 -> K = kc*32+16+half*8+(j-8).
  auto stage_async = [&](int bufIdx, int chunk) {
    const int ebase = chunk * E_CHUNK;
    #pragma unroll
    for (int it = 0; it < 8; ++it) {
      const int f   = it * 256 + t;   // 16B unit id, 0..2047
      const int row = f >> 5;         // codeword within chunk, 0..63
      const int uir = f & 31;         // 16B unit within 512B row
      const __bf16* src = cb_bf + (size_t)(ebase + row) * E_DIM + uir * 8;
      const int et = row >> 4, n = row & 15;
      const int kc = uir >> 2;
      const int hf = uir & 1;
      const int u  = (uir >> 1) & 1;
      const unsigned ldsOff =
          (unsigned)(uintptr_t)&bTile[bufIdx][et][kc][hf * 16 + n][u * 8];
      asm volatile("global_load_async_to_lds_b128 %0, %1, off"
                   :: "v"(ldsOff), "v"(src) : "memory");
    }
  };

  // ---- Phase 0: kick off chunk 0 async; stage cbnorm into LDS.
  stage_async(0, 0);
  {
    const f32x4 cn4 = *(const f32x4*)(cbnorm + t * 4);
    *(f32x4*)&cbnormS[t * 4] = cn4;
  }

  // ---- Phase 1: two A sub-tiles (rows wv*32+0..15 and +16..31) -> registers.
  // A 16x32 layout: lane m = lane&15; elems 0..7 -> K = kc*32+half*8+i,
  //                 elems 8..15 -> K = kc*32+16+half*8+i.
  const int m0 = wv * 32 + n16;          // sub-tile 0 row
  const int m1 = m0 + 16;                // sub-tile 1 row
  v16bf afrag0[K_CHUNKS], afrag1[K_CHUNKS];
  #pragma unroll
  for (int kc = 0; kc < K_CHUNKS; ++kc) {
    const int kb = kc * 32 + half * 8;
    #pragma unroll
    for (int i = 0; i < 8; ++i) {
      afrag0[kc][i]     = (__bf16)zplane[(size_t)(kb + i) * HW + m0];
      afrag0[kc][8 + i] = (__bf16)zplane[(size_t)(kb + 16 + i) * HW + m0];
      afrag1[kc][i]     = (__bf16)zplane[(size_t)(kb + i) * HW + m1];
      afrag1[kc][8 + i] = (__bf16)zplane[(size_t)(kb + 16 + i) * HW + m1];
    }
  }

  float bestD0[8], bestD1[8];
  int   bestI0[8], bestI1[8];
  #pragma unroll
  for (int r = 0; r < 8; ++r) {
    bestD0[r] = 3.4e38f; bestI0[r] = 0;
    bestD1[r] = 3.4e38f; bestI1[r] = 0;
  }

  wait_asynccnt0();      // chunk 0 resident (this wave's loads; barrier for rest)
  __syncthreads();

  // ---- Phase 2: sweep all 1024 codewords, double-buffered async pipeline.
  for (int ec = 0; ec < N_CHUNKS; ++ec) {
    const int cur = ec & 1;
    if (ec + 1 < N_CHUNKS)
      stage_async(cur ^ 1, ec + 1);    // overlap next chunk with compute

    const int ebase = ec * E_CHUNK;
    #pragma unroll
    for (int et = 0; et < E_CHUNK / 16; ++et) {
      v8f acc0 = {}, acc1 = {};
      #pragma unroll
      for (int kc = 0; kc < K_CHUNKS; ++kc) {
        const v16bf bfrag = *(const v16bf*)&bTile[cur][et][kc][lane][0];
        // one B fragment feeds two WMMAs (two row sub-tiles)
        acc0 = __builtin_amdgcn_wmma_f32_16x16x32_bf16(
            false, afrag0[kc], false, bfrag, (short)0, acc0, false, false);
        acc1 = __builtin_amdgcn_wmma_f32_16x16x32_bf16(
            false, afrag1[kc], false, bfrag, (short)0, acc1, false, false);
      }
      // D layout: acc[r] = D[row = half*8 + r][col e = n16]
      const int e  = ebase + et * 16 + n16;
      const float cn = cbnormS[e];               // LDS, no VMEM wait
      #pragma unroll
      for (int r = 0; r < 8; ++r) {
        const float d0 = fmaf(-2.0f, acc0[r], cn);
        if (d0 < bestD0[r]) { bestD0[r] = d0; bestI0[r] = e; }  // first-min wins
        const float d1 = fmaf(-2.0f, acc1[r], cn);
        if (d1 < bestD1[r]) { bestD1[r] = d1; bestI1[r] = e; }
      }
    }
    if (ec + 1 < N_CHUNKS)
      wait_asynccnt0();                // next chunk landed (own loads)
    __syncthreads();                   // all waves: done reading cur, next full
  }

  // ---- Cross-lane argmin inside each 16-lane group (xor<=8 stays in group).
  #pragma unroll
  for (int off = 8; off >= 1; off >>= 1) {
    #pragma unroll
    for (int r = 0; r < 8; ++r) {
      float d2; int i2;
      d2 = __shfl_xor(bestD0[r], off, 32);
      i2 = __shfl_xor(bestI0[r], off, 32);
      if (d2 < bestD0[r] || (d2 == bestD0[r] && i2 < bestI0[r])) {
        bestD0[r] = d2; bestI0[r] = i2;
      }
      d2 = __shfl_xor(bestD1[r], off, 32);
      i2 = __shfl_xor(bestI1[r], off, 32);
      if (d2 < bestD1[r] || (d2 == bestD1[r] && i2 < bestI1[r])) {
        bestD1[r] = d2; bestI1[r] = i2;
      }
    }
  }
  if (n16 == 0) {
    #pragma unroll
    for (int r = 0; r < 8; ++r) {
      idxArr[wv * 32 + half * 8 + r]      = bestI0[r];
      idxArr[wv * 32 + 16 + half * 8 + r] = bestI1[r];
    }
  }
  __syncthreads();

  // ---- Phase 3: exact fp32 gather -> NCHW store + fused loss reduction.
  const int row  = t;                      // one row per thread
  const int code = idxArr[row];
  const float* cbr  = cb_f32 + (size_t)code * E_DIM;
  const float* zp   = zplane + row;
  float*       outp = out + (size_t)b * E_DIM * HW + hwb + row;
  float accL = 0.f;
  #pragma unroll 4
  for (int c = 0; c < E_DIM; ++c) {
    const float q  = cbr[c];
    const float zv = zp[(size_t)c * HW];
    outp[(size_t)c * HW] = q;              // coalesced across 256 threads
    const float dlt = q - zv;
    accL = fmaf(dlt, dlt, accL);
  }
  #pragma unroll
  for (int off = 16; off >= 1; off >>= 1) accL += __shfl_down(accL, off, 32);
  if (lane == 0) redBuf[wv] = accL;
  __syncthreads();
  if (t == 0) {
    float tot = 0.f;
    #pragma unroll
    for (int i = 0; i < 8; ++i) tot += redBuf[i];
    // loss = (beta + 1) * mean = 1.25 * sum / 16777216
    atomicAdd(loss, tot * (1.25f / (float)OUT_ELEMS));
  }
}

// ---------------------------------------------------------------------------
extern "C" void kernel_launch(void* const* d_in, const int* in_sizes, int n_in,
                              void* d_out, int out_size, void* d_ws, size_t ws_size,
                              hipStream_t stream) {
  const float* z  = (const float*)d_in[0];   // 16*256*64*64
  const float* cb = (const float*)d_in[1];   // 1024*256
  float* out  = (float*)d_out;               // zq (16.7M) ++ loss (1)
  float* loss = out + OUT_ELEMS;

  __bf16* cb_bf  = (__bf16*)d_ws;                                   // 512 KB
  float*  cbnorm = (float*)((char*)d_ws + (size_t)N_E * E_DIM * 2); // 4 KB

  vq_prep<<<N_E, 256, 0, stream>>>(cb, cb_bf, cbnorm, loss);
  vq_main<<<OUT_ELEMS / E_DIM / M_TILE, 256, 0, stream>>>(
      z, cb, cb_bf, cbnorm, out, loss);
}
